// QweNet_49675591746193
// MI455X (gfx1250) — matrix operand order repaired and checked by reading, not compiled, not run on hardware
//
#include <hip/hip_runtime.h>
#include <hip/hip_bf16.h>

typedef __bf16 bf16_t;
typedef __attribute__((ext_vector_type(16))) __bf16 v16bf;
typedef __attribute__((ext_vector_type(8)))  __bf16 v8bf;
typedef __attribute__((ext_vector_type(8)))  float  v8f;

#define EPS 1e-5f

// ---------------------------------------------------------------------------
// WMMA helpers (CDNA5 v_wmma_f32_16x16x32_bf16)
// ---------------------------------------------------------------------------
__device__ __forceinline__ v8f wmma_bf16(v16bf a, v16bf b, v8f c) {
    return __builtin_amdgcn_wmma_f32_16x16x32_bf16(
        /*neg_a=*/false, a, /*neg_b=*/false, b,
        /*c_mod=*/(short)0, c, /*reuse_a=*/false, /*reuse_b=*/false);
}

// A fragment: 8 bf16 at p, 8 bf16 at p+16 (two 16B loads)
__device__ __forceinline__ v16bf load_fragA(const bf16_t* p) {
    v8bf lo = *(const v8bf*)(p);
    v8bf hi = *(const v8bf*)(p + 16);
    v16bf r;
#pragma unroll
    for (int i = 0; i < 8; ++i) { r[i] = lo[i]; r[i + 8] = hi[i]; }
    return r;
}

// B fragment: 8 bf16 at p, 8 bf16 at p+16 (two 16B loads) from pre-transposed Bt [N,K]
__device__ __forceinline__ v16bf load_fragB(const bf16_t* p) {
    v8bf lo = *(const v8bf*)(p);
    v8bf hi = *(const v8bf*)(p + 16);
    v16bf r;
#pragma unroll
    for (int i = 0; i < 8; ++i) { r[i] = lo[i]; r[i + 8] = hi[i]; }
    return r;
}

// ---------------------------------------------------------------------------
// GEMM (K == 128): C[M,N] = A[M,K](bf16,rm) * Bt[N,K](bf16,rm)^T (+bias)(+relu)
// One block per 128-row stripe; 8 waves; each wave owns 16 rows, preloads its
// 4 A fragments into VGPRs ONCE, then sweeps every 32-wide N tile streaming
// only the (L1/L2-resident) weight fragments.  A traffic = exactly one pass.
// HAS_BIAS / RELU are compile-time so the hot loop is branch-free.
// ---------------------------------------------------------------------------
template <bool HAS_BIAS, bool RELU>
__global__ void __launch_bounds__(256)
gemm_bf16_wmma_k128(const bf16_t* __restrict__ A, const bf16_t* __restrict__ Bt,
                    const float* __restrict__ bias, float* __restrict__ C,
                    int M, int N) {
    const int K = 128;
    const int wave = threadIdx.x >> 5;
    const int lane = threadIdx.x & 31;
    const int rowBase = blockIdx.x * 128 + wave * 16;
    if (rowBase >= M) return;

    int m = rowBase + (lane & 15);
    if (m >= M) m = M - 1;                 // clamp loads; stores guarded
    const int aoff = (lane & 16) ? 8  : 0;
    const int boff = (lane & 16) ? 8  : 0;
    const int nlane = lane & 15;
    const int rb = rowBase + ((lane & 16) ? 8 : 0);
    const bool fullTile = (rowBase + 16) <= M;   // hoisted store-guard test

    // preload all 4 A fragments (K=128) into registers
    const bf16_t* pA = A + (size_t)m * K + aoff;
    v16bf afrag[4];
#pragma unroll
    for (int ks = 0; ks < 4; ++ks) afrag[ks] = load_fragA(pA + ks * 32);

    for (int nb = 0; nb < N; nb += 32) {
        const int n0 = nb + nlane;
        const bf16_t* pB0 = Bt + (size_t)n0 * K + boff;
        const bf16_t* pB1 = Bt + (size_t)(n0 + 16) * K + boff;
        v8f acc0 = {}; v8f acc1 = {};
#pragma unroll
        for (int ks = 0; ks < 4; ++ks) {
            __builtin_prefetch(pB0 + ks * 32 + 2 * K, 0, 0);  // Bt row n0+2 (2*K elements ahead)
            acc0 = wmma_bf16(afrag[ks], load_fragB(pB0 + ks * 32), acc0);
            acc1 = wmma_bf16(afrag[ks], load_fragB(pB1 + ks * 32), acc1);
        }
        float bi0 = 0.0f, bi1 = 0.0f;
        if (HAS_BIAS) { bi0 = bias[n0]; bi1 = bias[n0 + 16]; }
        float* pc0 = C + (size_t)rb * N + n0;
        if (fullTile) {
#pragma unroll
            for (int r = 0; r < 8; ++r) {
                float v0 = acc0[r] + bi0;
                float v1 = acc1[r] + bi1;
                if (RELU) { v0 = fmaxf(v0, 0.0f); v1 = fmaxf(v1, 0.0f); }
                pc0[(size_t)r * N]      = v0;
                pc0[(size_t)r * N + 16] = v1;
            }
        } else {
#pragma unroll
            for (int r = 0; r < 8; ++r) {
                if (rb + r < M) {
                    float v0 = acc0[r] + bi0;
                    float v1 = acc1[r] + bi1;
                    if (RELU) { v0 = fmaxf(v0, 0.0f); v1 = fmaxf(v1, 0.0f); }
                    pc0[(size_t)r * N]      = v0;
                    pc0[(size_t)r * N + 16] = v1;
                }
            }
        }
    }
}

// ---------------------------------------------------------------------------
// Degree / symmetric norm (deg counts incoming edges + self loop)
// ---------------------------------------------------------------------------
__global__ void deg_init(float* deg, int n) {
    int i = blockIdx.x * blockDim.x + threadIdx.x;
    if (i < n) deg[i] = 1.0f;              // self loop
}
__global__ void deg_count(const int* __restrict__ dst, float* deg, int E) {
    int e = blockIdx.x * blockDim.x + threadIdx.x;
    if (e < E) atomicAdd(&deg[dst[e]], 1.0f);
}
__global__ void deg_rsqrt(float* dis, int n) {
    int i = blockIdx.x * blockDim.x + threadIdx.x;
    if (i < n) dis[i] = rsqrtf(dis[i]);
}

// ---------------------------------------------------------------------------
// GCN: init with self-loop term + bias, then edge scatter (L2-resident atomics)
// ---------------------------------------------------------------------------
__global__ void gcn_init(const float* __restrict__ hh, const float* __restrict__ dis,
                         const float* __restrict__ bg, float* __restrict__ c,
                         int M, int D_) {
    long i = (long)blockIdx.x * blockDim.x + threadIdx.x;
    if (i >= (long)M * D_) return;
    int row = (int)(i / D_), f = (int)(i % D_);
    float dn = dis[row];
    c[i] = hh[i] * dn * dn + bg[f];
}

__global__ void gcn_edge(const int* __restrict__ src, const int* __restrict__ dst,
                         const float* __restrict__ dis, const float* __restrict__ hh,
                         float* __restrict__ c, int E, int D_) {
    long t = (long)blockIdx.x * blockDim.x + threadIdx.x;
    const int per = D_ >> 2;               // float4 chunks per edge
    long e = t / per;
    if (e >= E) return;
    int f4 = (int)(t % per) << 2;
    int s = src[e], d = dst[e];
    float nrm = dis[s] * dis[d];
    float4 v = *(const float4*)(hh + (size_t)s * D_ + f4);
    float* cd = c + (size_t)d * D_ + f4;
    atomicAdd(cd + 0, v.x * nrm);
    atomicAdd(cd + 1, v.y * nrm);
    atomicAdd(cd + 2, v.z * nrm);
    atomicAdd(cd + 3, v.w * nrm);
}

// ---------------------------------------------------------------------------
// GRU gate math (gi/gh [M,3D] precomputed by WMMA GEMMs with biases)
// ---------------------------------------------------------------------------
__device__ __forceinline__ float sigmoidf(float x) { return 1.0f / (1.0f + expf(-x)); }

__global__ void gru_kernel(const float* __restrict__ gi, const float* __restrict__ gh,
                           const float* __restrict__ h, float* __restrict__ hn,
                           int M, int D_) {
    long i = (long)blockIdx.x * blockDim.x + threadIdx.x;
    if (i >= (long)M * D_) return;
    int row = (int)(i / D_), j = (int)(i % D_);
    size_t b = (size_t)row * 3 * D_;
    float r = sigmoidf(gi[b + j] + gh[b + j]);
    float z = sigmoidf(gi[b + D_ + j] + gh[b + D_ + j]);
    float nn = tanhf(gi[b + 2 * D_ + j] + r * gh[b + 2 * D_ + j]);
    hn[i] = (1.0f - z) * nn + z * h[i];
}

// ---------------------------------------------------------------------------
// BatchNorm (training mode, biased var): per-64-row partial column sums then
// atomic reduce; apply kernel fuses ReLU / f32+bf16 writeback / running max.
// ---------------------------------------------------------------------------
__global__ void colsum(const float* __restrict__ X, int M, int C_, float* st) {
    int col = threadIdx.x;                 // blockDim.x == C_
    int r0 = blockIdx.x * 64;
    int rend = min(r0 + 64, M);
    float s = 0.0f, q = 0.0f;
    for (int r = r0; r < rend; ++r) {
        float v = X[(size_t)r * C_ + col];
        s += v; q += v * v;
    }
    atomicAdd(&st[col], s);
    atomicAdd(&st[C_ + col], q);
}

// mode bits: 1=relu after BN, 2=write f32 (outF), 4=write bf16 (outB),
//            8=write mx, 16=max-accumulate into mx (else overwrite)
__global__ void bn_apply(const float* __restrict__ X, int M, int C_,
                         const float* __restrict__ st,
                         const float* __restrict__ gamma, const float* __restrict__ beta,
                         float* outF, bf16_t* outB, float* mx, int mode) {
    long i = (long)blockIdx.x * blockDim.x + threadIdx.x;
    if (i >= (long)M * C_) return;
    int col = (int)(i % C_);
    float invM = 1.0f / (float)M;
    float mean = st[col] * invM;
    float var  = st[C_ + col] * invM - mean * mean;
    float v = (X[i] - mean) * rsqrtf(var + EPS) * gamma[col] + beta[col];
    if (mode & 1) v = fmaxf(v, 0.0f);
    if (mode & 2) outF[i] = v;
    if (mode & 4) outB[i] = (bf16_t)v;
    if (mode & 8) mx[i] = (mode & 16) ? fmaxf(mx[i], v) : v;
}

// ---------------------------------------------------------------------------
// Weight conversion: plain f32->bf16, and transpose f32[K,N] -> bf16[N,K]
// ---------------------------------------------------------------------------
__global__ void cvt_bf16(const float* __restrict__ W, bf16_t* __restrict__ o, long n) {
    long i = (long)blockIdx.x * blockDim.x + threadIdx.x;
    if (i < n) o[i] = (bf16_t)W[i];
}
__global__ void cvtT_bf16(const float* __restrict__ W, bf16_t* __restrict__ o,
                          int K, int N) {
    long i = (long)blockIdx.x * blockDim.x + threadIdx.x;
    if (i >= (long)K * N) return;
    int k = (int)(i / N), n = (int)(i % N);
    o[(size_t)n * K + k] = (bf16_t)W[i];
}

// ---------------------------------------------------------------------------
// Final: out[i] = dot(Y[i,:], W2) + b2   (one wave32 per row)
// ---------------------------------------------------------------------------
__global__ void final_dot(const float* __restrict__ Y, const float* __restrict__ W2,
                          const float* __restrict__ b2, float* __restrict__ out,
                          int M, int H_) {
    int row = blockIdx.x;
    int lane = threadIdx.x;
    if (row >= M) return;
    float s = 0.0f;
    for (int k = lane; k < H_; k += 32)
        s += Y[(size_t)row * H_ + k] * W2[k];
#pragma unroll
    for (int off = 16; off > 0; off >>= 1)
        s += __shfl_down(s, off, 32);
    if (lane == 0) out[row] = s + b2[0];
}

// ---------------------------------------------------------------------------
// Host-side orchestration
// ---------------------------------------------------------------------------
static inline int cdiv(long a, long b) { return (int)((a + b - 1) / b); }

extern "C" void kernel_launch(void* const* d_in, const int* in_sizes, int n_in,
                              void* d_out, int out_size, void* d_ws, size_t ws_size,
                              hipStream_t stream) {
    const int F = 128, D = 128, H = 256, ITERS = 5;
    const int N = in_sizes[0] / F;
    const int E = in_sizes[1] / 2;

    const float* x    = (const float*)d_in[0];
    const int*   eidx = (const int*)d_in[1];
    const int*   src  = eidx;
    const int*   dst  = eidx + E;
    const float* W_in = (const float*)d_in[2];
    const float* b_in = (const float*)d_in[3];
    const float* g0   = (const float*)d_in[4];
    const float* be0  = (const float*)d_in[5];
    const float* W_g  = (const float*)d_in[6];
    const float* b_g  = (const float*)d_in[7];
    const float* W_ih = (const float*)d_in[8];
    const float* W_hh = (const float*)d_in[9];
    const float* b_ih = (const float*)d_in[10];
    const float* b_hh = (const float*)d_in[11];
    const float* g_c  = (const float*)d_in[12];
    const float* be_c = (const float*)d_in[13];
    const float* g_o  = (const float*)d_in[14];
    const float* be_o = (const float*)d_in[15];
    const float* W1   = (const float*)d_in[16];
    const float* b1   = (const float*)d_in[17];
    const float* g1   = (const float*)d_in[18];
    const float* be1  = (const float*)d_in[19];
    const float* W2   = (const float*)d_in[20];
    const float* b2   = (const float*)d_in[21];
    float* out = (float*)d_out;

    // ---- workspace carve ----
    char* p = (char*)d_ws;
    auto carve = [&](size_t bytes) -> char* {
        char* r = p;
        p += (bytes + 255) & ~(size_t)255;
        return r;
    };
    float*  dis    = (float*)carve((size_t)N * 4);
    bf16_t* x_bf   = (bf16_t*)carve((size_t)N * F * 2);
    bf16_t* h_bf   = (bf16_t*)carve((size_t)N * D * 2);
    bf16_t* c_bf   = (bf16_t*)carve((size_t)N * D * 2);
    bf16_t* mx_bf  = (bf16_t*)carve((size_t)N * D * 2);
    bf16_t* Wint   = (bf16_t*)carve((size_t)F * D * 2);  // W_in^T [D,F]
    bf16_t* Wgt    = (bf16_t*)carve((size_t)D * D * 2);  // W_g^T
    bf16_t* Wih_bf = (bf16_t*)carve((size_t)3 * D * D * 2); // already [3D,D]=Bt
    bf16_t* Whh_bf = (bf16_t*)carve((size_t)3 * D * D * 2);
    bf16_t* W1t    = (bf16_t*)carve((size_t)D * H * 2);  // W1^T [H,D]
    float*  hf32   = (float*)carve((size_t)N * D * 4);
    float*  hh     = (float*)carve((size_t)N * D * 4);
    float*  cbuf   = (float*)carve((size_t)N * D * 4);
    float*  hn     = (float*)carve((size_t)N * D * 4);
    float*  mx     = (float*)carve((size_t)N * D * 4);
    float*  gi     = (float*)carve((size_t)N * 3 * D * 4); // also reused as y [N,H]
    float*  gh     = (float*)carve((size_t)N * 3 * D * 4);
    float*  stats  = (float*)carve(2 * 256 * 4);
    float*  y      = gi;                                   // N*H <= N*3D

    const int T = 256;
    dim3 blk(T);

    // ---- degree / norm (constant across iterations) ----
    deg_init<<<cdiv(N, T), blk, 0, stream>>>(dis, N);
    deg_count<<<cdiv(E, T), blk, 0, stream>>>(dst, dis, E);
    deg_rsqrt<<<cdiv(N, T), blk, 0, stream>>>(dis, N);

    // ---- weight conversion (once per launch) ----
    cvtT_bf16<<<cdiv((long)F * D, T), blk, 0, stream>>>(W_in, Wint, F, D);
    cvtT_bf16<<<cdiv((long)D * D, T), blk, 0, stream>>>(W_g, Wgt, D, D);
    cvt_bf16<<<cdiv((long)3 * D * D, T), blk, 0, stream>>>(W_ih, Wih_bf, (long)3 * D * D);
    cvt_bf16<<<cdiv((long)3 * D * D, T), blk, 0, stream>>>(W_hh, Whh_bf, (long)3 * D * D);
    cvtT_bf16<<<cdiv((long)D * H, T), blk, 0, stream>>>(W1, W1t, D, H);
    cvt_bf16<<<cdiv((long)N * F, T), blk, 0, stream>>>(x, x_bf, (long)N * F);

    dim3 gemmBlk(256);
    const dim3 gemmGrid(cdiv(N, 128));
    const int ND = N * D;

    // ---- decoder input layer: relu(x@W_in + b_in) -> BN -> h, mx ----
    {
        gemm_bf16_wmma_k128<true, true><<<gemmGrid, gemmBlk, 0, stream>>>(
            x_bf, Wint, b_in, hn, N, D);
        hipMemsetAsync(stats, 0, 2 * D * 4, stream);
        colsum<<<cdiv(N, 64), dim3(D), 0, stream>>>(hn, N, D, stats);
        bn_apply<<<cdiv(ND, T), blk, 0, stream>>>(hn, N, D, stats, g0, be0,
                                                  hf32, h_bf, mx, 2 | 4 | 8);
    }

    // ---- 5 message-passing iterations ----
    for (int it = 0; it < ITERS; ++it) {
        // GCN: hh = h @ W_g ; c = scatter(norm * hh) + b_g
        gemm_bf16_wmma_k128<false, false><<<gemmGrid, gemmBlk, 0, stream>>>(
            h_bf, Wgt, nullptr, hh, N, D);
        gcn_init<<<cdiv(ND, T), blk, 0, stream>>>(hh, dis, b_g, cbuf, N, D);
        gcn_edge<<<cdiv((long)E * (D / 4), T), blk, 0, stream>>>(src, dst, dis, hh, cbuf, E, D);
        cvt_bf16<<<cdiv((long)ND, T), blk, 0, stream>>>(cbuf, c_bf, (long)ND);

        // GRU gates: gi = h@W_ih^T + b_ih ; gh = c@W_hh^T + b_hh
        gemm_bf16_wmma_k128<true, false><<<gemmGrid, gemmBlk, 0, stream>>>(
            h_bf, Wih_bf, b_ih, gi, N, 3 * D);
        gemm_bf16_wmma_k128<true, false><<<gemmGrid, gemmBlk, 0, stream>>>(
            c_bf, Whh_bf, b_hh, gh, N, 3 * D);
        gru_kernel<<<cdiv(ND, T), blk, 0, stream>>>(gi, gh, hf32, hn, N, D);

        // per-cycle BN; update h (f32 + bf16) and running max
        hipMemsetAsync(stats, 0, 2 * D * 4, stream);
        colsum<<<cdiv(N, 64), dim3(D), 0, stream>>>(hn, N, D, stats);
        bn_apply<<<cdiv(ND, T), blk, 0, stream>>>(hn, N, D, stats, g_c, be_c,
                                                  hf32, h_bf, mx, 2 | 4 | 8 | 16);
    }

    // ---- decoder output BN on mx -> bf16 for encoder GEMM ----
    hipMemsetAsync(stats, 0, 2 * D * 4, stream);
    colsum<<<cdiv(N, 64), dim3(D), 0, stream>>>(mx, N, D, stats);
    bn_apply<<<cdiv(ND, T), blk, 0, stream>>>(mx, N, D, stats, g_o, be_o,
                                              nullptr, mx_bf, nullptr, 4);

    // ---- encoder: relu(BN(mx@W1 + b1)) ----
    {
        gemm_bf16_wmma_k128<true, false><<<gemmGrid, gemmBlk, 0, stream>>>(
            mx_bf, W1t, b1, y, N, H);
        hipMemsetAsync(stats, 0, 2 * H * 4, stream);
        colsum<<<cdiv(N, 64), dim3(H), 0, stream>>>(y, N, H, stats);
        bn_apply<<<cdiv((long)N * H, T), blk, 0, stream>>>(y, N, H, stats, g1, be1,
                                                           y, nullptr, nullptr, 1 | 2);
    }

    // ---- final linear: [N,H] @ [H,1] + b2 ----
    final_dot<<<N, dim3(32), 0, stream>>>(y, W2, b2, out, N, H);
}
